// Duration_Predictor_52355651338556
// MI455X (gfx1250) — matrix operand order repaired and checked by reading, not compiled, not run on hardware
//
#include <hip/hip_runtime.h>

// ---------------- types ----------------
typedef __bf16 bf16_t;
typedef bf16_t v16bf __attribute__((ext_vector_type(16)));
typedef float  v8f   __attribute__((ext_vector_type(8)));
typedef unsigned int v8u __attribute__((ext_vector_type(8)));

#define C_CH 512
#define L_IN 2048
#define B_N  16
#define KSZ  5
#define NTOT (B_N * L_IN)          // 32768 columns
#define T_OUT (4 * L_IN)           // 8192
#define COEF_CONV 0.019764235376052372f   // 1/sqrt(512*5)
#define COEF_P    0.04419417382415922f    // 1/sqrt(512)
#define APACK_DW  (5 * 16 * 32 * 32 * 8)  // dwords per layer = 655360

__device__ __forceinline__ unsigned short f2bf(float f) {
  union { float f; unsigned u; } v; v.f = f;
  unsigned r = v.u + 0x7fffu + ((v.u >> 16) & 1u);
  return (unsigned short)(r >> 16);
}

// ---------------- weight pre-pack into A-fragment lane order ----------------
// A frag (16x32 bf16): lane=(half,m); VGPR j holds K=2j,2j+1 (+16*half).
// Index: [t][kblk][mblk][lane][j]
__global__ __launch_bounds__(256) void pack_w(const float* __restrict__ W,
                                              unsigned int* __restrict__ apack) {
  int idx = blockIdx.x * 256 + threadIdx.x;
  int j    = idx & 7;
  int lane = (idx >> 3) & 31;
  int mblk = (idx >> 8) & 31;
  int kblk = (idx >> 13) & 15;
  int t    = idx >> 17;
  if (t >= KSZ) return;
  int half = lane >> 4, m = lane & 15;
  int k0 = 2 * j + 16 * half;
  int co = mblk * 16 + m;
  int ci = kblk * 32 + k0;
  float w0 = W[((size_t)co * C_CH + ci) * KSZ + t] * COEF_CONV;
  float w1 = W[((size_t)co * C_CH + ci + 1) * KSZ + t] * COEF_CONV;
  apack[idx] = (unsigned)f2bf(w0) | ((unsigned)f2bf(w1) << 16);
}

// ---------------- encodings [B,C,L] f32 -> [B,L,C] bf16 (tiled transpose) ----
__global__ __launch_bounds__(256) void enc_to_bf16(const float* __restrict__ enc,
                                                   unsigned short* __restrict__ x) {
  __shared__ float tile[32][33];
  int b = blockIdx.z, c0 = blockIdx.y * 32, l0 = blockIdx.x * 32;
  int tx = threadIdx.x & 31, ty = threadIdx.x >> 5;
  const float* src = enc + ((size_t)b * C_CH + c0) * L_IN + l0;
#pragma unroll
  for (int r = 0; r < 4; ++r) tile[ty + 8 * r][tx] = src[(size_t)(ty + 8 * r) * L_IN + tx];
  __syncthreads();
#pragma unroll
  for (int r = 0; r < 4; ++r)
    x[((size_t)b * L_IN + l0 + ty + 8 * r) * C_CH + c0 + tx] = f2bf(tile[tx][ty + 8 * r]);
}

// ---------------- conv-as-GEMM with bf16 WMMA, f32 accumulate ----------------
// y[n][co] = sum_{t,ci} Wpacked * x[b, l+t-2, ci]  + bias[co]
// block: 8 waves (2M x 4N); wave tile M=32, N=64; K-steps: 5 taps x 16 ci-blocks
__global__ __launch_bounds__(256) void conv_wmma(const unsigned short* __restrict__ xin,
                                                 const unsigned int* __restrict__ apack,
                                                 const float* __restrict__ bias,
                                                 float* __restrict__ y) {
  const int lane = threadIdx.x & 31;
  const int wid  = threadIdx.x >> 5;
  const int wm = wid & 1, wn = wid >> 1;
  const int mBase = blockIdx.y * 64 + wm * 32;
  const int nBase = blockIdx.x * 256 + wn * 64;
  const int half = lane >> 4, nl = lane & 15;

  v8f acc[2][4];
  const v8f vzero = {0.f, 0.f, 0.f, 0.f, 0.f, 0.f, 0.f, 0.f};
#pragma unroll
  for (int mf = 0; mf < 2; ++mf)
#pragma unroll
    for (int nf = 0; nf < 4; ++nf) acc[mf][nf] = vzero;

  int col[4], lp[4];
#pragma unroll
  for (int nf = 0; nf < 4; ++nf) {
    col[nf] = nBase + nf * 16 + nl;
    lp[nf]  = col[nf] & (L_IN - 1);
  }

  for (int t = 0; t < KSZ; ++t) {
    size_t rowOff[4];
    bool ok[4];
#pragma unroll
    for (int nf = 0; nf < 4; ++nf) {
      int ls = lp[nf] + t - 2;
      ok[nf] = ((unsigned)ls < (unsigned)L_IN);
      rowOff[nf] = ok[nf] ? ((size_t)(col[nf] - lp[nf] + ls) * C_CH + (size_t)half * 16) : 0;
    }
    for (int kb = 0; kb < 16; ++kb) {
      const int ci0 = kb * 32;
      v16bf aF[2], bF[4];
#pragma unroll
      for (int mf = 0; mf < 2; ++mf) {
        const uint4* ap = (const uint4*)(apack +
            ((((size_t)t * 16 + kb) * 32 + (mBase >> 4) + mf) * 32 + lane) * 8);
        uint4 lo = ap[0], hi = ap[1];
        v8u u;
        u[0] = lo.x; u[1] = lo.y; u[2] = lo.z; u[3] = lo.w;
        u[4] = hi.x; u[5] = hi.y; u[6] = hi.z; u[7] = hi.w;
        aF[mf] = __builtin_bit_cast(v16bf, u);
      }
#pragma unroll
      for (int nf = 0; nf < 4; ++nf) {
        const uint4* bp = (const uint4*)(xin + rowOff[nf] + ci0);
        uint4 z; z.x = z.y = z.z = z.w = 0u;
        uint4 lo = ok[nf] ? bp[0] : z;
        uint4 hi = ok[nf] ? bp[1] : z;
        v8u u;
        u[0] = lo.x; u[1] = lo.y; u[2] = lo.z; u[3] = lo.w;
        u[4] = hi.x; u[5] = hi.y; u[6] = hi.z; u[7] = hi.w;
        bF[nf] = __builtin_bit_cast(v16bf, u);
      }
#pragma unroll
      for (int mf = 0; mf < 2; ++mf)
#pragma unroll
        for (int nf = 0; nf < 4; ++nf)
          acc[mf][nf] = __builtin_amdgcn_wmma_f32_16x16x32_bf16(
              false, aF[mf], false, bF[nf], (short)0, acc[mf][nf], false, false);
    }
  }

  // epilogue: D layout -> co = mBase + mf*16 + 8*half + r ; n = nBase + nf*16 + nl
  float bb[2][8];
#pragma unroll
  for (int mf = 0; mf < 2; ++mf)
#pragma unroll
    for (int r = 0; r < 8; ++r)
      bb[mf][r] = bias[mBase + mf * 16 + half * 8 + r];
#pragma unroll
  for (int mf = 0; mf < 2; ++mf)
#pragma unroll
    for (int nf = 0; nf < 4; ++nf) {
      size_t base = (size_t)(nBase + nf * 16 + nl) * C_CH + mBase + mf * 16 + half * 8;
      float4 s0, s1;
      s0.x = acc[mf][nf][0] + bb[mf][0]; s0.y = acc[mf][nf][1] + bb[mf][1];
      s0.z = acc[mf][nf][2] + bb[mf][2]; s0.w = acc[mf][nf][3] + bb[mf][3];
      s1.x = acc[mf][nf][4] + bb[mf][4]; s1.y = acc[mf][nf][5] + bb[mf][5];
      s1.z = acc[mf][nf][6] + bb[mf][6]; s1.w = acc[mf][nf][7] + bb[mf][7];
      float4* dst = (float4*)(y + base);
      dst[0] = s0; dst[1] = s1;
    }
}

// ---------------- LayerNorm(channel) + ReLU -> bf16 (one wave per column) ----
__global__ __launch_bounds__(256) void ln_relu(const float* __restrict__ y,
                                               const float* __restrict__ g,
                                               const float* __restrict__ be,
                                               unsigned short* __restrict__ xout) {
  int lane = threadIdx.x & 31, wid = threadIdx.x >> 5;
  size_t col = (size_t)blockIdx.x * 8 + wid;
  const float* p = y + col * C_CH + lane * 16;
  float v[16];
#pragma unroll
  for (int q = 0; q < 4; ++q) {
    float4 f = ((const float4*)p)[q];
    v[q * 4 + 0] = f.x; v[q * 4 + 1] = f.y; v[q * 4 + 2] = f.z; v[q * 4 + 3] = f.w;
  }
  float s = 0.f, sq = 0.f;
#pragma unroll
  for (int i = 0; i < 16; ++i) { s += v[i]; sq += v[i] * v[i]; }
#pragma unroll
  for (int d = 16; d > 0; d >>= 1) { s += __shfl_xor(s, d); sq += __shfl_xor(sq, d); }
  float mean = s * (1.0f / 512.0f);
  float var  = sq * (1.0f / 512.0f) - mean * mean;
  float rstd = rsqrtf(var + 1e-5f);
  unsigned o[8];
#pragma unroll
  for (int i = 0; i < 16; i += 2) {
    int c = lane * 16 + i;
    float a  = fmaxf((v[i]     - mean) * rstd * g[c]     + be[c],     0.0f);
    float b2 = fmaxf((v[i + 1] - mean) * rstd * g[c + 1] + be[c + 1], 0.0f);
    o[i >> 1] = (unsigned)f2bf(a) | ((unsigned)f2bf(b2) << 16);
  }
  uint4* dst = (uint4*)(xout + col * C_CH + lane * 16);
  uint4 d0; d0.x = o[0]; d0.y = o[1]; d0.z = o[2]; d0.w = o[3];
  uint4 d1; d1.x = o[4]; d1.y = o[5]; d1.z = o[6]; d1.w = o[7];
  dst[0] = d0; dst[1] = d1;
}

// ---------------- last LayerNorm fused with 1x1 projection + ReLU ----------
__global__ __launch_bounds__(256) void ln_proj(const float* __restrict__ y,
                                               const float* __restrict__ g,
                                               const float* __restrict__ be,
                                               const float* __restrict__ wp,
                                               const float* __restrict__ bp,
                                               float* __restrict__ pred) {
  int lane = threadIdx.x & 31, wid = threadIdx.x >> 5;
  size_t col = (size_t)blockIdx.x * 8 + wid;
  const float* p = y + col * C_CH + lane * 16;
  float v[16];
#pragma unroll
  for (int q = 0; q < 4; ++q) {
    float4 f = ((const float4*)p)[q];
    v[q * 4 + 0] = f.x; v[q * 4 + 1] = f.y; v[q * 4 + 2] = f.z; v[q * 4 + 3] = f.w;
  }
  float s = 0.f, sq = 0.f;
#pragma unroll
  for (int i = 0; i < 16; ++i) { s += v[i]; sq += v[i] * v[i]; }
#pragma unroll
  for (int d = 16; d > 0; d >>= 1) { s += __shfl_xor(s, d); sq += __shfl_xor(sq, d); }
  float mean = s * (1.0f / 512.0f);
  float var  = sq * (1.0f / 512.0f) - mean * mean;
  float rstd = rsqrtf(var + 1e-5f);
  float dotv = 0.f;
#pragma unroll
  for (int i = 0; i < 16; ++i) {
    int c = lane * 16 + i;
    float a = fmaxf((v[i] - mean) * rstd * g[c] + be[c], 0.0f);
    dotv += a * wp[c];
  }
#pragma unroll
  for (int d = 16; d > 0; d >>= 1) dotv += __shfl_xor(dotv, d);
  if (lane == 0) pred[col] = fmaxf(dotv * COEF_P + bp[0], 0.0f);
}

// ---------------- cumsum of durations (one wave per batch row) --------------
__global__ void scan_cum(const int* __restrict__ dur, int* __restrict__ cum) {
  int b = blockIdx.x, lane = threadIdx.x;
  int offset = 0;
  for (int c = 0; c < L_IN; c += 32) {
    int v = dur[b * L_IN + c + lane];
#pragma unroll
    for (int d = 1; d < 32; d <<= 1) {
      int n = __shfl_up(v, d);
      if (lane >= d) v += n;
    }
    cum[b * L_IN + c + lane] = v + offset;
    offset += __shfl(v, 31);
  }
}

// ---------------- length regulation: gather expand --------------------------
__global__ __launch_bounds__(256) void expand_gather(const float* __restrict__ enc,
                                                     const int* __restrict__ cum,
                                                     float* __restrict__ out) {
  __shared__ int sidx[256];
  int b = blockIdx.z, c0 = blockIdx.y * 32, t0 = blockIdx.x * 256;
  int tid = threadIdx.x;
  int t = t0 + tid;
  const int* cb = cum + b * L_IN;
  int lo = 0, hi = L_IN;
  while (lo < hi) {                         // searchsorted(cum, t, 'right')
    int mid = (lo + hi) >> 1;
    if (cb[mid] <= t) lo = mid + 1; else hi = mid;
  }
  sidx[tid] = lo;
  __syncthreads();
#pragma unroll 4
  for (int c = 0; c < 32; ++c) {
    size_t row = (size_t)b * C_CH + c0 + c;
    out[row * T_OUT + t0 + tid] = enc[row * L_IN + sidx[tid]];
  }
}

// ---------------- launch ----------------------------------------------------
extern "C" void kernel_launch(void* const* d_in, const int* in_sizes, int n_in,
                              void* d_out, int out_size, void* d_ws, size_t ws_size,
                              hipStream_t stream) {
  (void)in_sizes; (void)n_in; (void)out_size; (void)ws_size;
  const float* enc = (const float*)d_in[0];
  const int*   dur = (const int*)d_in[1];
  const float* W[3]  = {(const float*)d_in[2], (const float*)d_in[6],  (const float*)d_in[10]};
  const float* bc[3] = {(const float*)d_in[3], (const float*)d_in[7],  (const float*)d_in[11]};
  const float* gg[3] = {(const float*)d_in[4], (const float*)d_in[8],  (const float*)d_in[12]};
  const float* bb[3] = {(const float*)d_in[5], (const float*)d_in[9],  (const float*)d_in[13]};
  const float* Wp = (const float*)d_in[14];
  const float* bp = (const float*)d_in[15];

  char* ws = (char*)d_ws;
  unsigned short* xA = (unsigned short*)(ws);                                  // 32 MB
  unsigned short* xB = (unsigned short*)(ws + (size_t)32 * 1024 * 1024);       // 32 MB
  float* ybuf        = (float*)(ws + (size_t)64 * 1024 * 1024);                // 64 MB
  unsigned int* ap0  = (unsigned int*)(ws + (size_t)128 * 1024 * 1024);        // 2.5 MB x3
  unsigned int* ap1  = ap0 + APACK_DW;
  unsigned int* ap2  = ap1 + APACK_DW;
  int* cum           = (int*)(ws + (size_t)128 * 1024 * 1024 + (size_t)3 * APACK_DW * 4);

  float* expanded = (float*)d_out;
  float* pred = expanded + (size_t)B_N * C_CH * T_OUT;

  unsigned int* ap[3] = {ap0, ap1, ap2};
  for (int l = 0; l < 3; ++l)
    pack_w<<<APACK_DW / 256, 256, 0, stream>>>(W[l], ap[l]);

  enc_to_bf16<<<dim3(L_IN / 32, C_CH / 32, B_N), 256, 0, stream>>>(enc, xA);

  conv_wmma<<<dim3(NTOT / 256, C_CH / 64), 256, 0, stream>>>(xA, ap0, bc[0], ybuf);
  ln_relu<<<NTOT / 8, 256, 0, stream>>>(ybuf, gg[0], bb[0], xB);

  conv_wmma<<<dim3(NTOT / 256, C_CH / 64), 256, 0, stream>>>(xB, ap1, bc[1], ybuf);
  ln_relu<<<NTOT / 8, 256, 0, stream>>>(ybuf, gg[1], bb[1], xA);

  conv_wmma<<<dim3(NTOT / 256, C_CH / 64), 256, 0, stream>>>(xA, ap2, bc[2], ybuf);
  ln_proj<<<NTOT / 8, 256, 0, stream>>>(ybuf, gg[2], bb[2], Wp, bp, pred);

  scan_cum<<<B_N, 32, 0, stream>>>(dur, cum);
  expand_gather<<<dim3(T_OUT / 256, C_CH / 32, B_N), 256, 0, stream>>>(enc, cum, expanded);
}
